// GAT_WLN_78864189489700
// MI455X (gfx1250) — matrix operand order repaired and checked by reading, not compiled, not run on hardware
//
#include <hip/hip_runtime.h>
#include <hip/hip_bf16.h>

// GAT/WLN forward for MI455X (gfx1250, wave32, WMMA).
// GEMMs on v_wmma_f32_16x16x32_bf16 (f32 accumulate). Each wave computes a
// 16x32 output tile with TWO independent accumulators so back-to-back WMMAs
// never hit the D->C RAW hazard (no v_nop stalls). All K dims padded to
// multiples of 32 -> guard-free b128 loads. Scatter ops use f32 global
// atomics; the 335 MB pair output uses non-temporal stores.

namespace {

constexpr int HN = 256;   // hidden
constexpr int NN = 4096;  // nodes
constexpr int EE = 32768; // edges
constexpr int FF = 82;    // node features (padded to 96)
constexpr int DD = 6;     // edge features (padded to 32)
constexpr int FP = 96;    // padded F
constexpr int DP = 32;    // padded D

typedef __attribute__((ext_vector_type(16))) __bf16 v16bf;
typedef __attribute__((ext_vector_type(8)))  float  v8f;
typedef __attribute__((ext_vector_type(4)))  float  v4f;

enum { A_PLAIN = 0, A_CONCAT = 1, A_MUL = 2, A_EDGE = 3 };
enum { O_STORE = 0, O_SCATTER = 1 };

__device__ __forceinline__ v4f ld4(const float* p) {
  return *(const v4f*)p;   // 16B-aligned by construction -> global_load_b128
}

__device__ __forceinline__ void pack4(v16bf& d, int o, v4f s) {
#pragma unroll
  for (int i = 0; i < 4; ++i) d[o + i] = (__bf16)s[i];
}

// One K=32 step over a 16x32 output tile: shared A fragment, two B fragments,
// two independent WMMA accumulator chains. Layouts used here (k = 16*hi + s for A and B alike):
//  A (16x32 bf16): lane's 16 elems = 16 contiguous floats at ra[kA..]
//  B (32x16 bf16): lane's 16 elems = 16 contiguous floats at rb[kB..]
template <bool MUL>
__device__ __forceinline__ void mma_step2(v8f& c0, v8f& c1,
                                          const float* __restrict__ ra,
                                          const float* __restrict__ rm,
                                          const float* __restrict__ rb0,
                                          const float* __restrict__ rb1,
                                          int kA, int kB)
{
  v4f a0 = ld4(ra + kA),      a1 = ld4(ra + kA + 4);
  v4f a2 = ld4(ra + kA + 8), a3 = ld4(ra + kA + 12);
  if (MUL) {
    a0 *= ld4(rm + kA);      a1 *= ld4(rm + kA + 4);
    a2 *= ld4(rm + kA + 8); a3 *= ld4(rm + kA + 12);
  }
  v4f p0 = ld4(rb0 + kB), p1 = ld4(rb0 + kB + 4);
  v4f p2 = ld4(rb0 + kB + 8), p3 = ld4(rb0 + kB + 12);
  v4f q0 = ld4(rb1 + kB), q1 = ld4(rb1 + kB + 4);
  v4f q2 = ld4(rb1 + kB + 8), q3 = ld4(rb1 + kB + 12);

  v16bf a, b0, b1;
  pack4(a, 0, a0);  pack4(a, 4, a1);  pack4(a, 8, a2);  pack4(a, 12, a3);
  pack4(b0, 0, p0); pack4(b0, 4, p1); pack4(b0, 8, p2); pack4(b0, 12, p3);
  pack4(b1, 0, q0); pack4(b1, 4, q1); pack4(b1, 8, q2); pack4(b1, 12, q3);

  c0 = __builtin_amdgcn_wmma_f32_16x16x32_bf16(false, a, false, b0,
                                               (short)0, c0, false, false);
  c1 = __builtin_amdgcn_wmma_f32_16x16x32_bf16(false, a, false, b1,
                                               (short)0, c1, false, false);
}

// ---------------------------------------------------------------------------
// Generic WMMA GEMM: out[m][n] = act( sum_k A[m][k] * W[n][k] + bias[n] )
// N (output cols) fixed = HN = 256; one wave per 16x32 tile; K compile-time.
// ---------------------------------------------------------------------------
template <int AM, int OM, int RELU, int BIAS, int K, int K0, int A1S>
__global__ __launch_bounds__(256) void gemm_wmma(
    const float* __restrict__ A0, const float* __restrict__ A1,
    const float* __restrict__ W, const float* __restrict__ bias,
    const int* __restrict__ srcIdx, const int* __restrict__ dstIdx,
    float* __restrict__ out, int M)
{
  const int wid  = threadIdx.x >> 5;
  const int lane = threadIdx.x & 31;
  const int tile = blockIdx.x * 8 + wid;   // (M/16)*8 tiles total
  const int m0   = (tile >> 3) * 16;       // 8 N-tile-pairs (HN=256, 32 wide)
  const int n0   = (tile & 7) * 32;
  if (m0 >= M) return;

  // A fragment row: lanes l and l+16 both map to M = m0 + (l&15).
  const int am = m0 + (lane & 15);
  const float* arow0 = nullptr;
  const float* arow1 = nullptr;   // pre-shifted by -K0 for concat/edge
  if (AM == A_PLAIN) {
    arow0 = A0 + (size_t)am * K;
  } else if (AM == A_CONCAT || AM == A_EDGE) {
    const int r0 = (AM == A_EDGE) ? srcIdx[am] : am;
    arow0 = A0 + (size_t)r0 * K0;
    arow1 = A1 + (size_t)am * A1S - K0;
  } else {  // A_MUL
    arow0 = A0 + (size_t)am * K;
    arow1 = A1 + (size_t)am * K;
  }

  // B: out = A @ W.T, so B[k][n] = W[n][k]; n fixed per lane; two N-tiles.
  const float* wrow0 = W + (size_t)(n0 + (lane & 15)) * K;
  const float* wrow1 = wrow0 + (size_t)16 * K;

  const int hi = lane >> 4;
  const int kA = hi * 16;    // A half-wave K offset
  const int kB = hi * 16;    // B half-wave K offset

  v8f c0 = {}, c1 = {};
  if (AM == A_CONCAT || AM == A_EDGE) {
    for (int kb = 0; kb < K0; kb += 32)
      mma_step2<false>(c0, c1, arow0 + kb, nullptr, wrow0 + kb, wrow1 + kb,
                       kA, kB);
    for (int kb = K0; kb < K; kb += 32)
      mma_step2<false>(c0, c1, arow1 + kb, nullptr, wrow0 + kb, wrow1 + kb,
                       kA, kB);
  } else if (AM == A_MUL) {
    for (int kb = 0; kb < K; kb += 32)
      mma_step2<true>(c0, c1, arow0 + kb, arow1 + kb, wrow0 + kb, wrow1 + kb,
                      kA, kB);
  } else {
    for (int kb = 0; kb < K; kb += 32)
      mma_step2<false>(c0, c1, arow0 + kb, nullptr, wrow0 + kb, wrow1 + kb,
                       kA, kB);
  }

  // C/D layout: VGPR r -> row m0 + r + 8*hi, col n0 + (lane&15) (+16 for c1).
  const int n = n0 + (lane & 15);
  const float bv0 = BIAS ? bias[n] : 0.f;
  const float bv1 = BIAS ? bias[n + 16] : 0.f;
#pragma unroll
  for (int r = 0; r < 8; ++r) {
    const int m = m0 + r + 8 * hi;
    float v0 = c0[r] + bv0;
    float v1 = c1[r] + bv1;
    if (RELU) { v0 = v0 > 0.f ? v0 : 0.f; v1 = v1 > 0.f ? v1 : 0.f; }
    if (OM == O_STORE) {
      out[(size_t)m * HN + n]      = v0;
      out[(size_t)m * HN + n + 16] = v1;
    } else {
      const size_t drow = (size_t)dstIdx[m] * HN;       // segment_sum
      atomicAdd(out + drow + n, v0);
      atomicAdd(out + drow + n + 16, v1);
    }
  }
}

// ---------------------------------------------------------------------------
__global__ __launch_bounds__(256) void zero_k(float* p, int n) {
  int i = blockIdx.x * 256 + threadIdx.x;
  if (i < n) p[i] = 0.f;
}

// Row-pad a [M,Ks] matrix to [M,Kd] with zeros (Kd multiple of 32).
__global__ __launch_bounds__(256) void pad_k(const float* __restrict__ s,
                                             float* __restrict__ d,
                                             int Ks, int Kd, int total) {
  const int i = blockIdx.x * 256 + threadIdx.x;
  if (i >= total) return;
  const int m = i / Kd, k = i - m * Kd;
  d[i] = (k < Ks) ? s[(size_t)m * Ks + k] : 0.f;
}

// msg2 = hW3[src] * (edge_attr @ W2.T + b2); scatter-add into agg2[dst].
__global__ __launch_bounds__(256) void wl2_edge(
    const float* __restrict__ hW3, const float* __restrict__ ea,
    const float* __restrict__ W2, const float* __restrict__ b2,
    const int* __restrict__ src, const int* __restrict__ dst,
    float* __restrict__ agg2)
{
  const int e = blockIdx.x, n = threadIdx.x;
  const int s = src[e], d = dst[e];
  const float* er = ea + (size_t)e * DD;
  const float* wr = W2 + (size_t)n * DD;
  float t = b2[n];
#pragma unroll
  for (int k = 0; k < DD; ++k) t += er[k] * wr[k];
  atomicAdd(agg2 + (size_t)d * HN + n, hW3[(size_t)s * HN + n] * t);
}

// a_s[m] = g[m].asrc, a_d[m] = g[m].adst  (block per node, LDS reduce)
__global__ __launch_bounds__(256) void att_scores(
    const float* __restrict__ g, const float* __restrict__ asrc,
    const float* __restrict__ adst, float* __restrict__ a_s,
    float* __restrict__ a_d)
{
  __shared__ float sh[2][256];
  const int m = blockIdx.x, n = threadIdx.x;
  const float gv = g[(size_t)m * HN + n];
  sh[0][n] = gv * asrc[n];
  sh[1][n] = gv * adst[n];
  __syncthreads();
  for (int s2 = 128; s2 > 0; s2 >>= 1) {
    if (n < s2) { sh[0][n] += sh[0][n + s2]; sh[1][n] += sh[1][n + s2]; }
    __syncthreads();
  }
  if (n == 0) { a_s[m] = sh[0][0]; a_d[m] = sh[1][0]; }
}

__device__ __forceinline__ unsigned ford(float f) {  // order-preserving map
  unsigned b = __float_as_uint(f);
  return (b & 0x80000000u) ? ~b : (b | 0x80000000u);
}
__device__ __forceinline__ float funord(unsigned o) {
  return __uint_as_float((o & 0x80000000u) ? (o & 0x7fffffffu) : ~o);
}

__global__ __launch_bounds__(256) void att_max(
    const float* __restrict__ a_s, const float* __restrict__ a_d,
    const int* __restrict__ src, const int* __restrict__ dst,
    unsigned* __restrict__ emax)
{
  const int i = blockIdx.x * 256 + threadIdx.x;
  if (i >= EE + NN) return;
  const int s = i < EE ? src[i] : i - EE;
  const int d = i < EE ? dst[i] : i - EE;
  float e = a_s[s] + a_d[d];
  e = e > 0.f ? e : 0.2f * e;                    // leaky_relu
  atomicMax(emax + d, ford(e));                  // segment_max
}

__global__ __launch_bounds__(256) void att_exp(
    const float* __restrict__ a_s, const float* __restrict__ a_d,
    const int* __restrict__ src, const int* __restrict__ dst,
    const unsigned* __restrict__ emax, float* __restrict__ ex,
    float* __restrict__ esum)
{
  const int i = blockIdx.x * 256 + threadIdx.x;
  if (i >= EE + NN) return;
  const int s = i < EE ? src[i] : i - EE;
  const int d = i < EE ? dst[i] : i - EE;
  float e = a_s[s] + a_d[d];
  e = e > 0.f ? e : 0.2f * e;
  const float x = expf(e - funord(emax[d]));
  ex[i] = x;
  atomicAdd(esum + d, x);
}

__global__ __launch_bounds__(256) void att_glob(
    const float* __restrict__ g, const int* __restrict__ src,
    const int* __restrict__ dst, const float* __restrict__ ex,
    const float* __restrict__ esum, float* __restrict__ glob)
{
  const int i = blockIdx.x, n = threadIdx.x;
  const int s = i < EE ? src[i] : i - EE;
  const int d = i < EE ? dst[i] : i - EE;
  const float alpha = ex[i] / esum[d];
  atomicAdd(glob + (size_t)d * HN + n, alpha * g[(size_t)s * HN + n]);
}

__global__ __launch_bounds__(256) void add3_k(
    const float* __restrict__ a, const float* __restrict__ b,
    const float* __restrict__ bias, float* __restrict__ o, int n)
{
  const int i = blockIdx.x * 256 + threadIdx.x;
  if (i < n) o[i] = a[i] + b[i] + bias[i & (HN - 1)];
}

// q[m][c] = sum_n t[m][n] * W3[c][n]  (block per node)
__global__ __launch_bounds__(256) void q_k(
    const float* __restrict__ t, const float* __restrict__ W3,
    float* __restrict__ q)
{
  __shared__ float sh[256];
  const int m = blockIdx.x, n = threadIdx.x;
  const float tv = t[(size_t)m * HN + n];
  for (int c = 0; c < 5; ++c) {
    sh[n] = tv * W3[c * HN + n];
    __syncthreads();
    for (int s2 = 128; s2 > 0; s2 >>= 1) {
      if (n < s2) sh[n] += sh[n + s2];
      __syncthreads();
    }
    if (n == 0) q[m * 5 + c] = sh[0];
    __syncthreads();
  }
}

// out[(i*N+j)*5 + c] = q[i][c]+q[j][c]; rows i==j -> -1. NT stores (335 MB).
__global__ __launch_bounds__(256) void pair_k(
    const float* __restrict__ q, float* __restrict__ out)
{
  const int j = blockIdx.x * 256 + threadIdx.x;
  const int i = blockIdx.y;
  const size_t base = ((size_t)i * NN + j) * 5;
  if (i == j) {
#pragma unroll
    for (int c = 0; c < 5; ++c)
      __builtin_nontemporal_store(-1.0f, out + base + c);
  } else {
    const float* qi = q + (size_t)i * 5;
    const float* qj = q + (size_t)j * 5;
#pragma unroll
    for (int c = 0; c < 5; ++c)
      __builtin_nontemporal_store(qi[c] + qj[c], out + base + c);
  }
}

} // namespace

extern "C" void kernel_launch(void* const* d_in, const int* in_sizes, int n_in,
                              void* d_out, int out_size, void* d_ws,
                              size_t ws_size, hipStream_t stream)
{
  const float* x        = (const float*)d_in[0];
  const int*   ei       = (const int*)  d_in[1];
  const float* edge_attr= (const float*)d_in[2];
  const float* W_lin    = (const float*)d_in[3];
  const float* W_lin2   = (const float*)d_in[4];
  const float* W_lin3   = (const float*)d_in[5];
  const float* wl1_W1   = (const float*)d_in[6];
  const float* wl1_b1   = (const float*)d_in[7];
  const float* wl1_W2   = (const float*)d_in[8];
  const float* wl1_b2   = (const float*)d_in[9];
  const float* wl2_W2   = (const float*)d_in[10];
  const float* wl2_b2   = (const float*)d_in[11];
  const float* wl2_W3   = (const float*)d_in[12];
  const float* wl2_b3   = (const float*)d_in[13];
  const float* gat_W    = (const float*)d_in[14];
  const float* gat_asrc = (const float*)d_in[15];
  const float* gat_adst = (const float*)d_in[16];
  const float* gat_b    = (const float*)d_in[17];

  const int* src = ei;
  const int* dst = ei + EE;

  // Workspace layout (floats). Total ~12.1M floats ~= 48.5 MB.
  float* ws = (float*)d_ws;
  const size_t NH = (size_t)NN * HN;  // 1,048,576
  float* h    = ws;
  float* h2   = ws + 1 * NH;
  float* agg  = ws + 2 * NH;
  float* hW3  = ws + 3 * NH;
  float* agg2 = ws + 4 * NH;
  float* loc  = ws + 5 * NH;
  float* g    = ws + 6 * NH;
  float* glob = ws + 7 * NH;
  float* u    = ws + 8 * NH;
  float* t    = ws + 9 * NH;
  float* a_s  = ws + 10 * NH;
  float* a_d  = a_s + NN;
  unsigned* emax = (unsigned*)(a_d + NN);
  float* esum = (float*)(emax + NN);
  float* ex   = esum + NN;
  float* q    = ex + (EE + NN);
  float* xp   = q + 5 * NN;                 // padded x      [NN, FP]
  float* Wlp  = xp + (size_t)NN * FP;       // padded W_lin  [HN, FP]
  float* W1p  = Wlp + (size_t)HN * FP;      // padded wl1_W1 [HN, 288]
  float* eap  = W1p + (size_t)HN * 288;     // padded edges  [EE, DP]
  float* outp = (float*)d_out;

  const dim3 blk(256);
  const unsigned zb = (unsigned)((NH + 255) / 256);

  // Zero accumulators every call (graph replays reuse the workspace).
  zero_k<<<dim3(zb), blk, 0, stream>>>(agg,  (int)NH);
  zero_k<<<dim3(zb), blk, 0, stream>>>(agg2, (int)NH);
  zero_k<<<dim3(zb), blk, 0, stream>>>(glob, (int)NH);
  zero_k<<<dim3((2 * NN + 255) / 256), blk, 0, stream>>>((float*)emax, 2 * NN);

  // Pad K dims to multiples of 32 so MMA loops are guard-free + b128 loads.
  pad_k<<<dim3((NN * FP + 255) / 256), blk, 0, stream>>>(x, xp, FF, FP, NN * FP);
  pad_k<<<dim3((HN * FP + 255) / 256), blk, 0, stream>>>(W_lin, Wlp, FF, FP,
                                                         HN * FP);
  pad_k<<<dim3((HN * 288 + 255) / 256), blk, 0, stream>>>(wl1_W1, W1p,
                                                          HN + DD, 288,
                                                          HN * 288);
  pad_k<<<dim3((EE * DP + 255) / 256), blk, 0, stream>>>(edge_attr, eap,
                                                         DD, DP, EE * DP);

  // 1) h = relu(x @ W_lin.T)                     [4096,96]x[256,96]
  gemm_wmma<A_PLAIN, O_STORE, 1, 0, FP, 0, 0>
      <<<dim3(NN / 16), blk, 0, stream>>>(
      xp, nullptr, Wlp, nullptr, nullptr, nullptr, h, NN);
  // 2) msg = relu([h[src]|ea] @ W1.T + b1); agg[dst] += msg  (32768x256x288)
  gemm_wmma<A_EDGE, O_SCATTER, 1, 1, 288, HN, DP>
      <<<dim3(EE / 16), blk, 0, stream>>>(
      h, eap, W1p, wl1_b1, src, dst, agg, EE);
  // 3) h2 = relu([agg|h] @ W2.T + b2)            (4096x256x512)
  gemm_wmma<A_CONCAT, O_STORE, 1, 1, 512, HN, HN>
      <<<dim3(NN / 16), blk, 0, stream>>>(
      agg, h, wl1_W2, wl1_b2, nullptr, nullptr, h2, NN);
  // 4) hW3 = h2 @ wl2_W3.T + b3
  gemm_wmma<A_PLAIN, O_STORE, 0, 1, HN, 0, 0>
      <<<dim3(NN / 16), blk, 0, stream>>>(
      h2, nullptr, wl2_W3, wl2_b3, nullptr, nullptr, hW3, NN);
  // 5) agg2[dst] += hW3[src] * (ea @ wl2_W2.T + b2)
  wl2_edge<<<dim3(EE), blk, 0, stream>>>(hW3, edge_attr, wl2_W2, wl2_b2,
                                         src, dst, agg2);
  // 6) local = (agg2 * h2) @ wl2_W3.T + b3
  gemm_wmma<A_MUL, O_STORE, 0, 1, HN, 0, 0>
      <<<dim3(NN / 16), blk, 0, stream>>>(
      agg2, h2, wl2_W3, wl2_b3, nullptr, nullptr, loc, NN);
  // 7) g = h2 @ gat_W.T
  gemm_wmma<A_PLAIN, O_STORE, 0, 0, HN, 0, 0>
      <<<dim3(NN / 16), blk, 0, stream>>>(
      h2, nullptr, gat_W, nullptr, nullptr, nullptr, g, NN);
  // 8-11) GAT attention: scores, segment max, softmax, weighted aggregate
  att_scores<<<dim3(NN), blk, 0, stream>>>(g, gat_asrc, gat_adst, a_s, a_d);
  att_max<<<dim3((EE + NN + 255) / 256), blk, 0, stream>>>(a_s, a_d, src, dst,
                                                           emax);
  att_exp<<<dim3((EE + NN + 255) / 256), blk, 0, stream>>>(a_s, a_d, src, dst,
                                                           emax, ex, esum);
  att_glob<<<dim3(EE + NN), blk, 0, stream>>>(g, src, dst, ex, esum, glob);
  // 12) u = local + glob + gat_b
  add3_k<<<dim3(zb), blk, 0, stream>>>(loc, glob, gat_b, u, (int)NH);
  // 13) t = u @ W_lin2.T
  gemm_wmma<A_PLAIN, O_STORE, 0, 0, HN, 0, 0>
      <<<dim3(NN / 16), blk, 0, stream>>>(
      u, nullptr, W_lin2, nullptr, nullptr, nullptr, t, NN);
  // 14) q = t @ W_lin3.T
  q_k<<<dim3(NN), blk, 0, stream>>>(t, W_lin3, q);
  // 15) pair expansion (335 MB NT stores, bandwidth-bound floor)
  pair_k<<<dim3(NN / 256, NN), blk, 0, stream>>>(q, outp);
}
